// OctFormerBlock_9010841387718
// MI455X (gfx1250) — matrix operand order repaired and checked by reading, not compiled, not run on hardware
//
#include <hip/hip_runtime.h>
#include <math.h>

// ---------------- problem constants ----------------
#define NTOK   131072
#define CCH    256
#define NHEAD  8
#define KWIN   32
#define NWIN   (NTOK / KWIN)       // 4096
#define DHEAD  (CCH / NHEAD)       // 32
#define POSB   25
#define RPENUM 51
#define ATTN_SCALE 0.17677669529663687f   // 32^-0.5
#define LN_EPS 1e-5f

// ---------------- WMMA / vector types ----------------
typedef __attribute__((ext_vector_type(16))) __bf16 v16bf;
typedef __attribute__((ext_vector_type(8)))  __bf16 v8bf;
typedef __attribute__((ext_vector_type(8)))  float  v8f;
typedef __attribute__((ext_vector_type(4)))  int    v4i;

#define SHUF16(lo, hi8) __builtin_shufflevector(lo, hi8, 0,1,2,3,4,5,6,7,8,9,10,11,12,13,14,15)

// gfx1250 async global->LDS path (guarded so compilation never regresses)
#if defined(__gfx1250__) && __has_builtin(__builtin_amdgcn_global_load_async_to_lds_b128) && \
    __has_builtin(__builtin_amdgcn_s_wait_asynccnt)
#define GEMM_ASYNC 1
#else
#define GEMM_ASYNC 0
#endif
#define AS1 __attribute__((address_space(1)))
#define AS3 __attribute__((address_space(3)))

// =====================================================================
// f32 -> bf16 conversion (weights)
// =====================================================================
__global__ void cvt_f32_bf16_kernel(const float* __restrict__ in,
                                    __bf16* __restrict__ out, int n) {
    int i = blockIdx.x * 256 + threadIdx.x;
    if (i < n) out[i] = (__bf16)in[i];
}

// =====================================================================
// LayerNorm over C=256, output bf16. 256 threads = 8 waves, 1 row/wave.
// =====================================================================
__global__ __launch_bounds__(256)
void layernorm_bf16_kernel(const float* __restrict__ x,
                           const float* __restrict__ g,
                           const float* __restrict__ b,
                           __bf16* __restrict__ out) {
    int lane = threadIdx.x & 31;
    int wave = threadIdx.x >> 5;
    int row  = blockIdx.x * 8 + wave;
    const float* xr = x + (size_t)row * CCH;

    float v[8];
    float s = 0.f;
#pragma unroll
    for (int i = 0; i < 8; ++i) { v[i] = xr[lane * 8 + i]; s += v[i]; }
#pragma unroll
    for (int off = 16; off > 0; off >>= 1) s += __shfl_xor(s, off, 32);
    float mean = s * (1.0f / CCH);
    float vs = 0.f;
#pragma unroll
    for (int i = 0; i < 8; ++i) { float d = v[i] - mean; vs += d * d; }
#pragma unroll
    for (int off = 16; off > 0; off >>= 1) vs += __shfl_xor(vs, off, 32);
    float inv = rsqrtf(vs * (1.0f / CCH) + LN_EPS);

    __bf16* orow = out + (size_t)row * CCH;
#pragma unroll
    for (int i = 0; i < 8; ++i) {
        int c = lane * 8 + i;
        orow[c] = (__bf16)((v[i] - mean) * inv * g[c] + b[c]);
    }
}

// =====================================================================
// Tiled bf16 GEMM, f32 accumulate, v_wmma_f32_16x16x32_bf16.
// Block tile 128x64, BK=64, double-buffered LDS, A-tile staged with
// GLOBAL_LOAD_ASYNC_TO_LDS_B128 (ASYNCcnt-tracked) when available.
// 8 waves (4x2), each wave a 32x32 tile; 8 WMMA per K-stage.
// Epilogues: 0 = +bias -> bf16 ; 1 = +bias,GELU -> bf16 ; 2 = +bias+res -> f32
// =====================================================================
enum { EPI_BIAS = 0, EPI_GELU = 1, EPI_RES = 2 };

template <int MODE>
__global__ __launch_bounds__(256)
void gemm_bf16_kernel(const __bf16* __restrict__ A,   // [M, Kc] row-major
                      const __bf16* __restrict__ B,   // [Kc, Nc] row-major
                      const float*  __restrict__ bias,
                      const float*  __restrict__ res,
                      __bf16* __restrict__ out_bf,
                      float*  __restrict__ out_f,
                      int Nc, int Kc) {
    __shared__ __attribute__((aligned(32))) __bf16 sA[2][128 * 64];  // [m][k]
    __shared__ __attribute__((aligned(32))) __bf16 sB[2][64 * 64];   // transposed [n][k]

    int tid  = threadIdx.x;
    int bn   = blockIdx.x * 64;
    int bm   = blockIdx.y * 128;
    int lane = tid & 31, wave = tid >> 5;
    int wm = wave >> 1, wn = wave & 1;
    int m0 = wm * 32, n0 = wn * 32;
    int l15 = lane & 15, hi = lane >> 4;

    v8f acc[2][2] = {};

    // stage loader: A tile 128x64 (4 async b128 per thread), B tile 64x64
    // stored transposed so B-fragments are contiguous-K LDS reads.
    auto stage_load = [&](int kk, int p) {
#pragma unroll
        for (int i = 0; i < 4; ++i) {
            int c = tid + i * 256;
            int row = c >> 3, col8 = (c & 7) * 8;
#if GEMM_ASYNC
            __builtin_amdgcn_global_load_async_to_lds_b128(
                (AS1 v4i*)&A[(size_t)(bm + row) * Kc + kk + col8],
                (AS3 v4i*)&sA[p][row * 64 + col8], 0, 0);
#else
            *(uint4*)&sA[p][row * 64 + col8] =
                *(const uint4*)&A[(size_t)(bm + row) * Kc + kk + col8];
#endif
        }
#pragma unroll
        for (int i = 0; i < 2; ++i) {
            int c = tid + i * 256;
            int krow = c >> 3, col8 = (c & 7) * 8;
            const __bf16* src = &B[(size_t)(kk + krow) * Nc + bn + col8];
#pragma unroll
            for (int e = 0; e < 8; ++e) sB[p][(col8 + e) * 64 + krow] = src[e];
        }
    };

    const int S = Kc >> 6;   // K stages of 64
    int buf = 0;
    stage_load(0, 0);

    for (int s = 0; s < S; ++s) {
        if (s + 1 < S) stage_load((s + 1) << 6, buf ^ 1);
#if GEMM_ASYNC
        if (s + 1 < S) __builtin_amdgcn_s_wait_asynccnt(4);  // this stage's 4 done
        else           __builtin_amdgcn_s_wait_asynccnt(0);
#endif
        __syncthreads();   // stage `buf` (async A + DS-stored B) visible to all waves

#pragma unroll
        for (int ks = 0; ks < 2; ++ks) {
            // A frag: lane row m=l15; elems e: k = hi*16+e (contiguous)
            v16bf af[2], bfr[2];
#pragma unroll
            for (int ti = 0; ti < 2; ++ti) {
                const __bf16* ap = &sA[buf][(m0 + ti * 16 + l15) * 64 + ks * 32];
                af[ti] = *(const v16bf*)&ap[hi * 16];
            }
            // B frag: lane col n=l15; elems e: k = hi*16+e (contiguous in sB)
#pragma unroll
            for (int tj = 0; tj < 2; ++tj)
                bfr[tj] = *(const v16bf*)&sB[buf][(n0 + tj * 16 + l15) * 64 + ks * 32 + hi * 16];

#pragma unroll
            for (int ti = 0; ti < 2; ++ti)
#pragma unroll
                for (int tj = 0; tj < 2; ++tj)
                    acc[ti][tj] = __builtin_amdgcn_wmma_f32_16x16x32_bf16(
                        false, af[ti], false, bfr[tj], (short)0, acc[ti][tj],
                        false, false);
        }
        __syncthreads();   // compute on `buf` done before next stage overwrites buf^1's twin
        buf ^= 1;
    }

    // epilogue: C layout m = ti*16 + hi*8 + r, n = tj*16 + l15
#pragma unroll
    for (int ti = 0; ti < 2; ++ti)
#pragma unroll
        for (int tj = 0; tj < 2; ++tj)
#pragma unroll
            for (int r = 0; r < 8; ++r) {
                int row = bm + m0 + ti * 16 + hi * 8 + r;
                int col = bn + n0 + tj * 16 + l15;
                size_t idx = (size_t)row * Nc + col;
                float v = acc[ti][tj][r] + bias[col];
                if (MODE == EPI_GELU)
                    v = 0.5f * v * (1.0f + erff(v * 0.70710678118654752f));
                if (MODE == EPI_RES)
                    out_f[idx] = v + res[idx];
                else
                    out_bf[idx] = (__bf16)v;
            }
}

// =====================================================================
// Windowed attention: 1 block per window (NW=4096), wave = head (H=8).
// QK^T and P*V via WMMA; scale+RPE+mask+softmax fused on accumulators.
// qkv layout: [N, 768], cols = s*256 + h*32 + d (s: q/k/v).
// =====================================================================
__global__ __launch_bounds__(256)
void attn_kernel(const __bf16* __restrict__ qkv,
                 const int*    __restrict__ rel_pos,   // [NW,32,32,3]
                 const float*  __restrict__ mask,      // [NW,32,32]
                 const float*  __restrict__ rpe_table, // [3*51, 8]
                 __bf16* __restrict__ out) {           // [N, 256]
    __shared__ __attribute__((aligned(32))) __bf16 sV[NHEAD][KWIN * DHEAD]; // V^T [d][j]
    __shared__ __attribute__((aligned(32))) __bf16 sP[NHEAD][KWIN * KWIN];  // P [t][j]

    int w    = blockIdx.x;
    int tid  = threadIdx.x;
    int lane = tid & 31;
    int h    = tid >> 5;
    int l15  = lane & 15, hi = lane >> 4;
    size_t wbase = (size_t)w * KWIN;

    // stage V transposed: lane = token j
    {
        const __bf16* vsrc = &qkv[(wbase + lane) * 768 + 512 + h * 32];
#pragma unroll
        for (int d = 0; d < DHEAD; ++d) sV[h][d * 32 + lane] = vsrc[d];
    }

    // ---- Q @ K^T (fragments straight from global; layouts are contiguous) ----
    v16bf qf[2], kf[2];
#pragma unroll
    for (int ti = 0; ti < 2; ++ti) {
        const __bf16* qp = &qkv[(wbase + ti * 16 + l15) * 768 + h * 32];
        qf[ti] = *(const v16bf*)&qp[hi * 16];
    }
#pragma unroll
    for (int tj = 0; tj < 2; ++tj)   // B frag = rows of K matrix (contiguous)
        kf[tj] = *(const v16bf*)&qkv[(wbase + tj * 16 + l15) * 768 + 256 + h * 32 + hi * 16];

    v8f acc[2][2] = {};
#pragma unroll
    for (int ti = 0; ti < 2; ++ti)
#pragma unroll
        for (int tj = 0; tj < 2; ++tj)
            acc[ti][tj] = __builtin_amdgcn_wmma_f32_16x16x32_bf16(
                false, qf[ti], false, kf[tj], (short)0, acc[ti][tj], false, false);

    // ---- scale + RPE + mask ----
#pragma unroll
    for (int ti = 0; ti < 2; ++ti)
#pragma unroll
        for (int tj = 0; tj < 2; ++tj)
#pragma unroll
            for (int r = 0; r < 8; ++r) {
                int t = ti * 16 + hi * 8 + r;
                int j = tj * 16 + l15;
                size_t e = (wbase + t) * KWIN + j;
                const int* rp = &rel_pos[e * 3];
                float rpe = 0.f;
#pragma unroll
                for (int a = 0; a < 3; ++a) {
                    int p = rp[a];
                    p = p < -POSB ? -POSB : (p > POSB ? POSB : p);
                    rpe += rpe_table[(p + POSB + a * RPENUM) * NHEAD + h];
                }
                acc[ti][tj][r] = acc[ti][tj][r] * ATTN_SCALE + rpe + mask[e];
            }

    // ---- softmax over j (row = 16 lanes of one half x 2 tj tiles) ----
#pragma unroll
    for (int ti = 0; ti < 2; ++ti)
#pragma unroll
        for (int r = 0; r < 8; ++r) {
            float v0 = acc[ti][0][r], v1 = acc[ti][1][r];
            float m = fmaxf(v0, v1);
#pragma unroll
            for (int off = 1; off <= 8; off <<= 1) m = fmaxf(m, __shfl_xor(m, off, 32));
            float e0 = __expf(v0 - m), e1 = __expf(v1 - m);
            float s = e0 + e1;
#pragma unroll
            for (int off = 1; off <= 8; off <<= 1) s += __shfl_xor(s, off, 32);
            float inv = 1.0f / s;
            int t = ti * 16 + hi * 8 + r;
            sP[h][t * 32 + l15]      = (__bf16)(e0 * inv);
            sP[h][t * 32 + 16 + l15] = (__bf16)(e1 * inv);
        }
    __syncthreads();

    // ---- P @ V ----
    v16bf pf[2], vf[2];
#pragma unroll
    for (int ti = 0; ti < 2; ++ti) {
        const __bf16* pp = &sP[h][(ti * 16 + l15) * 32];
        pf[ti] = *(const v16bf*)&pp[hi * 16];
    }
#pragma unroll
    for (int tj = 0; tj < 2; ++tj)   // B frag from V^T: contiguous j
        vf[tj] = *(const v16bf*)&sV[h][(tj * 16 + l15) * 32 + hi * 16];

    v8f oacc[2][2] = {};
#pragma unroll
    for (int ti = 0; ti < 2; ++ti)
#pragma unroll
        for (int tj = 0; tj < 2; ++tj)
            oacc[ti][tj] = __builtin_amdgcn_wmma_f32_16x16x32_bf16(
                false, pf[ti], false, vf[tj], (short)0, oacc[ti][tj], false, false);

    // store: token t, channel h*32 + d
#pragma unroll
    for (int ti = 0; ti < 2; ++ti)
#pragma unroll
        for (int tj = 0; tj < 2; ++tj)
#pragma unroll
            for (int r = 0; r < 8; ++r) {
                int t = ti * 16 + hi * 8 + r;
                int d = tj * 16 + l15;
                out[(wbase + t) * CCH + h * 32 + d] = (__bf16)oacc[ti][tj][r];
            }
}

// =====================================================================
// host launch
// =====================================================================
extern "C" void kernel_launch(void* const* d_in, const int* in_sizes, int n_in,
                              void* d_out, int out_size, void* d_ws, size_t ws_size,
                              hipStream_t stream) {
    const float* data      = (const float*)d_in[0];
    const int*   rel_pos   = (const int*)  d_in[1];
    const float* mask      = (const float*)d_in[2];
    const float* w_qkv     = (const float*)d_in[3];
    const float* b_qkv     = (const float*)d_in[4];
    const float* w_proj    = (const float*)d_in[5];
    const float* b_proj    = (const float*)d_in[6];
    const float* rpe_table = (const float*)d_in[7];
    const float* ln1_g     = (const float*)d_in[8];
    const float* ln1_b     = (const float*)d_in[9];
    const float* ln2_g     = (const float*)d_in[10];
    const float* ln2_b     = (const float*)d_in[11];
    const float* w_fc1     = (const float*)d_in[12];
    const float* b_fc1     = (const float*)d_in[13];
    const float* w_fc2     = (const float*)d_in[14];
    const float* b_fc2     = (const float*)d_in[15];
    float* out = (float*)d_out;

    // workspace carve-up (FFN buffer overlays dead qkv+attn regions)
    char* ws = (char*)d_ws;
    size_t off = 0;
    auto carve = [&](size_t bytes) {
        char* p = ws + off;
        off += (bytes + 255) & ~(size_t)255;
        return p;
    };
    __bf16* big     = (__bf16*)carve((size_t)NTOK * 1024 * sizeof(__bf16)); // 268 MB
    __bf16* qkv_b   = big;                                  // [N,768]
    __bf16* attnout = big + (size_t)NTOK * 768;             // [N,256]
    __bf16* ffn     = big;                                  // [N,1024] (reuse)
    __bf16* hbuf    = (__bf16*)carve((size_t)NTOK * CCH * sizeof(__bf16));
    float*  data2   = (float*) carve((size_t)NTOK * CCH * sizeof(float));
    __bf16* wq = (__bf16*)carve(256 * 768  * sizeof(__bf16));
    __bf16* wp = (__bf16*)carve(256 * 256  * sizeof(__bf16));
    __bf16* w1 = (__bf16*)carve(256 * 1024 * sizeof(__bf16));
    __bf16* w2 = (__bf16*)carve(1024 * 256 * sizeof(__bf16));

    // weight conversions
    cvt_f32_bf16_kernel<<<(256 * 768 + 255) / 256, 256, 0, stream>>>(w_qkv, wq, 256 * 768);
    cvt_f32_bf16_kernel<<<(256 * 256 + 255) / 256, 256, 0, stream>>>(w_proj, wp, 256 * 256);
    cvt_f32_bf16_kernel<<<(256 * 1024 + 255) / 256, 256, 0, stream>>>(w_fc1, w1, 256 * 1024);
    cvt_f32_bf16_kernel<<<(1024 * 256 + 255) / 256, 256, 0, stream>>>(w_fc2, w2, 1024 * 256);

    // LN1 -> h
    layernorm_bf16_kernel<<<NTOK / 8, 256, 0, stream>>>(data, ln1_g, ln1_b, hbuf);
    // QKV GEMM (+bias)
    gemm_bf16_kernel<EPI_BIAS><<<dim3(768 / 64, NTOK / 128), 256, 0, stream>>>(
        hbuf, wq, b_qkv, nullptr, qkv_b, nullptr, 768, 256);
    // attention
    attn_kernel<<<NWIN, 256, 0, stream>>>(qkv_b, rel_pos, mask, rpe_table, attnout);
    // proj GEMM (+bias, +residual data) -> data2 (f32)
    gemm_bf16_kernel<EPI_RES><<<dim3(256 / 64, NTOK / 128), 256, 0, stream>>>(
        attnout, wp, b_proj, data, nullptr, data2, 256, 256);
    // LN2 -> h2 (reuse hbuf)
    layernorm_bf16_kernel<<<NTOK / 8, 256, 0, stream>>>(data2, ln2_g, ln2_b, hbuf);
    // FC1 GEMM (+bias, GELU) -> ffn (bf16)
    gemm_bf16_kernel<EPI_GELU><<<dim3(1024 / 64, NTOK / 128), 256, 0, stream>>>(
        hbuf, w1, b_fc1, nullptr, ffn, nullptr, 1024, 256);
    // FC2 GEMM (+bias, +residual data2) -> out (f32)
    gemm_bf16_kernel<EPI_RES><<<dim3(256 / 64, NTOK / 128), 256, 0, stream>>>(
        ffn, w2, b_fc2, data2, nullptr, out, 256, 1024);
}
